// ForwardEncoder_47957604827685
// MI455X (gfx1250) — compile-verified, matrix operand order checked by reading, not run on hardware
//
#include <hip/hip_runtime.h>
#include <hip/hip_bf16.h>

typedef __bf16 bf16_t;
typedef __attribute__((ext_vector_type(16))) __bf16 bf16x16;
typedef __attribute__((ext_vector_type(8)))  __bf16 bf16x8;
typedef __attribute__((ext_vector_type(8)))  float   f32x8;

#define NROWS 1024   // B*S
#define DMODEL 512
#define DIN 1024
#define NHEAD 8
#define DH 128
#define SEQ 128
#define NBATCH 8

// WMMA 16-bit fragment element mapping (ISA 7.12.2):
//   lane = hf*16 + r ; element e <-> k = 8*hf + (e&7) + 16*(e>=8)
// Inverse: hf = (k>>3)&1 ; e = (k&7) | ((k>>4)<<3)
// Fragment-major LDS layout: [laneIdx][hf][e] -> each lane's 16 operand
// elements are 32 contiguous bytes (2x ds_load_b128).

__device__ __forceinline__ bf16x8 pack8(float4 u, float4 w) {
    bf16x8 p;
    p[0] = (bf16_t)u.x; p[1] = (bf16_t)u.y; p[2] = (bf16_t)u.z; p[3] = (bf16_t)u.w;
    p[4] = (bf16_t)w.x; p[5] = (bf16_t)w.y; p[6] = (bf16_t)w.z; p[7] = (bf16_t)w.w;
    return p;
}

__device__ __forceinline__ float fast_tanh(float x) {
    float t = __expf(-2.f * fabsf(x));          // v_exp_f32
    return copysignf((1.f - t) / (1.f + t), x);
}
__device__ __forceinline__ float fast_silu(float x) {
    return x / (1.f + __expf(-x));
}

// ---------------------------------------------------------------------------
// Tiled GEMM: C = act(alpha * A(f32,MxK) @ B(f32,KxN) + bias + res)
// Requires K%32==0 and N%64==0 (true for every call site). Only M is guarded.
// Block tile 64x64, 256 threads = 8 waves; per wave one 16-row strip and two
// 16x16 column tiles of v_wmma_f32_16x16x32_bf16 (f32 accumulate).
// Software-pipelined: next k-tile is prefetched to registers while the
// current tile's WMMA issues.
// ---------------------------------------------------------------------------
__global__ __launch_bounds__(256)
void wmma_gemm_kernel(const float* __restrict__ A, int lda,
                      const float* __restrict__ Bw, int ldb,
                      const float* __restrict__ bias,
                      const float* __restrict__ res, int ldr,
                      float* __restrict__ C, int ldc,
                      int M, int N, int K, float alpha, int act) {
    __shared__ bf16_t AsF[64 * 2 * 16];   // ((row*2 + hf)*16 + e)
    __shared__ bf16_t BsF[64 * 2 * 16];   // ((col*2 + hf)*16 + e)
    const int tid  = threadIdx.x;
    const int m0   = blockIdx.y * 64;
    const int n0   = blockIdx.x * 64;
    const int wave = tid >> 5;
    const int lane = tid & 31;
    const int r    = lane & 15;
    const int hf   = lane >> 4;
    const int tm   = (wave & 3) * 16;
    const int tn   = (wave >> 2) * 32;

    // A staging: thread -> (row, 8-wide k block)
    const int a_row = tid >> 2;
    const int a_k0  = (tid & 3) << 3;
    const int a_hf  = (a_k0 >> 3) & 1;
    const int a_e0  = (a_k0 >> 4) << 3;
    const bool a_ok = (m0 + a_row) < M;
    const float* a_base = A + (size_t)(m0 + a_row) * lda + a_k0;
    // B staging: thread -> (col, 8-wide k block)
    const int b_col = tid & 63;
    const int b_k0  = (tid >> 6) << 3;
    const int b_hf  = (b_k0 >> 3) & 1;
    const int b_e0  = (b_k0 >> 4) << 3;
    const float* b_base = Bw + (size_t)b_k0 * ldb + n0 + b_col;

    f32x8 acc0, acc1;
#pragma unroll
    for (int j = 0; j < 8; ++j) { acc0[j] = 0.f; acc1[j] = 0.f; }

    // ---- prologue prefetch (k-tile 0) ----
    bf16x8 apk, bpk;
    {
        if (a_ok) apk = pack8(*(const float4*)a_base, *(const float4*)(a_base + 4));
        else { for (int e = 0; e < 8; ++e) apk[e] = (bf16_t)0.f; }
        const float* bp = b_base;
#pragma unroll
        for (int i = 0; i < 8; ++i) bpk[i] = (bf16_t)bp[(size_t)i * ldb];
    }

    for (int kk = 0; kk < K; kk += 32) {
        __syncthreads();   // previous tile's fragment reads done
        *(bf16x8*)&AsF[((a_row << 1) + a_hf) * 16 + a_e0] = apk;
        *(bf16x8*)&BsF[((b_col << 1) + b_hf) * 16 + b_e0] = bpk;
        __syncthreads();

        if (kk + 32 < K) {         // prefetch next tile while WMMA runs
            if (a_ok) {
                const float* ap = a_base + kk + 32;
                apk = pack8(*(const float4*)ap, *(const float4*)(ap + 4));
            }
            const float* bp = b_base + (size_t)(kk + 32) * ldb;
#pragma unroll
            for (int i = 0; i < 8; ++i) bpk[i] = (bf16_t)bp[(size_t)i * ldb];
        }

        bf16x16 af  = *(const bf16x16*)&AsF[((tm + r) * 2 + hf) * 16];
        bf16x16 bf0 = *(const bf16x16*)&BsF[((tn + r) * 2 + hf) * 16];
        bf16x16 bf1 = *(const bf16x16*)&BsF[((tn + 16 + r) * 2 + hf) * 16];
        acc0 = __builtin_amdgcn_wmma_f32_16x16x32_bf16(false, af, false, bf0,
                                                       (short)0, acc0, false, false);
        acc1 = __builtin_amdgcn_wmma_f32_16x16x32_bf16(false, af, false, bf1,
                                                       (short)0, acc1, false, false);
    }

    // C/D layout: row = 8*hf + vgpr_j, col = lane%16
#pragma unroll
    for (int j = 0; j < 8; ++j) {
        int row = m0 + tm + (hf << 3) + j;
        if (row >= M) continue;
#pragma unroll
        for (int cn = 0; cn < 2; ++cn) {
            int col = n0 + tn + cn * 16 + r;
            float v = alpha * (cn ? acc1[j] : acc0[j]);
            if (bias) v += bias[col];
            if (res)  v += res[(size_t)row * ldr + col];
            if (act == 1) v = fast_tanh(v);
            C[(size_t)row * ldc + col] = v;
        }
    }
}

// ---------------------------------------------------------------------------
// LayerNorm over last dim (one row per block)
// ---------------------------------------------------------------------------
__global__ __launch_bounds__(256)
void ln_kernel(const float* __restrict__ x, const float* __restrict__ g,
               const float* __restrict__ b, float* __restrict__ y, int Dn) {
    const int row = blockIdx.x, tid = threadIdx.x;
    __shared__ float red[256];
    float s = 0.f;
    for (int c = tid; c < Dn; c += 256) s += x[(size_t)row * Dn + c];
    red[tid] = s; __syncthreads();
    for (int st = 128; st > 0; st >>= 1) { if (tid < st) red[tid] += red[tid + st]; __syncthreads(); }
    float mu = red[0] / Dn; __syncthreads();
    float s2 = 0.f;
    for (int c = tid; c < Dn; c += 256) { float d = x[(size_t)row * Dn + c] - mu; s2 += d * d; }
    red[tid] = s2; __syncthreads();
    for (int st = 128; st > 0; st >>= 1) { if (tid < st) red[tid] += red[tid + st]; __syncthreads(); }
    float inv = rsqrtf(red[0] / Dn + 1e-5f); __syncthreads();
    for (int c = tid; c < Dn; c += 256)
        y[(size_t)row * Dn + c] = (x[(size_t)row * Dn + c] - mu) * inv * g[c] + b[c];
}

// ---------------------------------------------------------------------------
// Causal conv (K=4) + SiLU : xm is a strided view of xz (ld 2048)
// ---------------------------------------------------------------------------
__global__ void conv_silu_kernel(const float* __restrict__ xz,
                                 const float* __restrict__ cw,
                                 const float* __restrict__ cb,
                                 float* __restrict__ xc) {
    int i = blockIdx.x * blockDim.x + threadIdx.x;    // NROWS*DIN
    int row = i >> 10, c = i & 1023;
    int s = row & (SEQ - 1);
    float acc = cb[c];
#pragma unroll
    for (int j = 0; j < 4; ++j) {
        int sj = s - 3 + j;
        if (sj >= 0) acc += xz[(size_t)(row - 3 + j) * 2048 + c] * cw[j * DIN + c];
    }
    xc[i] = fast_silu(acc);
}

// ---------------------------------------------------------------------------
// Gate logits: ip = xc@W_i + b_i ; fp = log_sigmoid(xc@W_f + b_f)
// One wave per (row, head)
// ---------------------------------------------------------------------------
__global__ __launch_bounds__(256)
void gates_kernel(const float* __restrict__ xc,
                  const float* __restrict__ Wi, const float* __restrict__ bi,
                  const float* __restrict__ Wf, const float* __restrict__ bfp,
                  float* __restrict__ ipb, float* __restrict__ fpb) {
    int gw = (blockIdx.x * blockDim.x + threadIdx.x) >> 5;
    int lane = threadIdx.x & 31;
    int row = gw >> 3, h = gw & 7;
    float si = 0.f, sf = 0.f;
    for (int d = lane; d < DIN; d += 32) {
        float xv = xc[(size_t)row * DIN + d];
        si += xv * Wi[d * NHEAD + h];
        sf += xv * Wf[d * NHEAD + h];
    }
    for (int off = 16; off > 0; off >>= 1) {
        si += __shfl_xor(si, off);
        sf += __shfl_xor(sf, off);
    }
    if (lane == 0) {
        ipb[row * NHEAD + h] = si + bi[h];
        float xf = sf + bfp[h];
        fpb[row * NHEAD + h] = (xf >= 0.f) ? -log1pf(__expf(-xf))
                                           : xf - log1pf(__expf(xf));
    }
}

// ---------------------------------------------------------------------------
// Full-sequence chunk-parallel mLSTM: one block per (batch, head).
// S = Q K^T (WMMA), causal decay weights, den = max(|rowsum|, e^{-m}),
// H = (W (.) S) V (WMMA), h = H / den.
// Q/K/V/Sw staged in LDS in fragment-major layout -> ds_load_b128 operands.
// ---------------------------------------------------------------------------
#define FIDX(li, kk32, hfv) ((((li) * 4 + (kk32)) * 2 + (hfv)) * 16)

__global__ __launch_bounds__(256)
void mlstm_scan_kernel(const float* __restrict__ q, const float* __restrict__ k,
                       const float* __restrict__ v,
                       const float* __restrict__ ipb, const float* __restrict__ fpb,
                       float* __restrict__ hb) {
    __shared__ bf16_t QF[SEQ * 8 * 16];   // A-frag layout by row s, k=d ; reused as SwF (k=s)
    __shared__ bf16_t KF[SEQ * 8 * 16];   // B-frag layout by col s', k=d
    __shared__ bf16_t VF[DH * 8 * 16];    // B-frag layout by col d, k=s
    __shared__ float gsh[SEQ], pmx[SEQ], emn[SEQ], rden[SEQ];

    const int b = blockIdx.x >> 3, h = blockIdx.x & 7;
    const int tid  = threadIdx.x;
    const int wave = tid >> 5, lane = tid & 31;
    const int r = lane & 15, hf = lane >> 4;
    const int wrow = wave << 4;
    const int rowbase = b * SEQ;

    // Stage Q,K: thread -> (row s, 8-wide d block); coalesced float4 loads.
    for (int c = tid; c < SEQ * 16; c += 256) {
        int s  = c >> 4;
        int d0 = (c & 15) << 3;
        int kk32 = d0 >> 5, w0 = d0 & 31;
        int hfv = (w0 >> 3) & 1, e0 = (w0 >> 4) << 3;
        size_t g = (size_t)(rowbase + s) * DIN + (h << 7) + d0;
        *(bf16x8*)&QF[FIDX(s, kk32, hfv) + e0] =
            pack8(*(const float4*)(q + g), *(const float4*)(q + g + 4));
        *(bf16x8*)&KF[FIDX(s, kk32, hfv) + e0] =
            pack8(*(const float4*)(k + g), *(const float4*)(k + g + 4));
    }
    // Stage V: thread -> (8-wide s block, col d); lanes coalesce over d.
    for (int c = tid; c < 16 * DH; c += 256) {
        int sb = c >> 7;             // 0..15
        int d  = c & 127;
        int s0 = sb << 3;
        int kk32 = s0 >> 5, w0 = s0 & 31;
        int hfv = (w0 >> 3) & 1, e0 = (w0 >> 4) << 3;
        bf16x8 pk;
#pragma unroll
        for (int i = 0; i < 8; ++i)
            pk[i] = (bf16_t)v[(size_t)(rowbase + s0 + i) * DIN + (h << 7) + d];
        *(bf16x8*)&VF[FIDX(d, kk32, hfv) + e0] = pk;
    }
    if (tid == 0) {
        float a = 0.f, pm = -1e30f;
        for (int s = 0; s < SEQ; ++s) {
            a += fpb[(rowbase + s) * NHEAD + h];
            float gg = ipb[(rowbase + s) * NHEAD + h] - a;   // i_s - a_s
            gsh[s] = gg;
            pm = fmaxf(pm, gg);
            pmx[s] = pm;                    // m_t = a_t + pmax_t
            emn[s] = __expf(-(a + pm));     // exp(-m_t)
        }
    }
    __syncthreads();

    // ---- S = Q K^T  (M=N=128, K=DH=128) ----
    f32x8 acc[8];
    for (int tc = 0; tc < 8; ++tc)
        for (int j = 0; j < 8; ++j) acc[tc][j] = 0.f;
#pragma unroll
    for (int kk32 = 0; kk32 < 4; ++kk32) {
        bf16x16 af = *(const bf16x16*)&QF[FIDX(wrow + r, kk32, hf)];
#pragma unroll
        for (int tc = 0; tc < 8; ++tc) {
            bf16x16 bfr = *(const bf16x16*)&KF[FIDX(tc * 16 + r, kk32, hf)];
            acc[tc] = __builtin_amdgcn_wmma_f32_16x16x32_bf16(false, af, false, bfr,
                                                              (short)0, acc[tc], false, false);
        }
    }

    // ---- causal decay weights + row sums; SwF aliases QF (k dim = s) ----
    float rs[8];
#pragma unroll
    for (int j = 0; j < 8; ++j) rs[j] = 0.f;
    for (int tc = 0; tc < 8; ++tc) {
        int s = tc * 16 + r;
        int kk32s = s >> 5, w0 = s & 31;
        int hfs = (w0 >> 3) & 1;
        int es  = (w0 & 7) | ((w0 >> 4) << 3);
        float gv = gsh[s];
#pragma unroll
        for (int j = 0; j < 8; ++j) {
            int t = wrow + (hf << 3) + j;
            float val = acc[tc][j];
            float w = (s <= t) ? __expf(gv - pmx[t]) : 0.f;   // arg <= 0, no overflow
            val *= w;
            QF[FIDX(t, kk32s, hfs) + es] = (bf16_t)val;
            rs[j] += val;
        }
    }
#pragma unroll
    for (int j = 0; j < 8; ++j) {
        float vsum = rs[j];
        for (int off = 1; off < 16; off <<= 1) vsum += __shfl_xor(vsum, off);
        if (r == 0) {
            int t = wrow + (hf << 3) + j;
            rden[t] = 1.f / fmaxf(fabsf(vsum), emn[t]);   // reciprocal once per row
        }
    }
    __syncthreads();

    // ---- H = Sw @ V  (M=128, N=DH=128, K=SEQ=128) ----
    f32x8 hacc[8];
    for (int tc = 0; tc < 8; ++tc)
        for (int j = 0; j < 8; ++j) hacc[tc][j] = 0.f;
#pragma unroll
    for (int kk32 = 0; kk32 < 4; ++kk32) {
        bf16x16 af = *(const bf16x16*)&QF[FIDX(wrow + r, kk32, hf)];
#pragma unroll
        for (int tc = 0; tc < 8; ++tc) {
            bf16x16 bfr = *(const bf16x16*)&VF[FIDX(tc * 16 + r, kk32, hf)];
            hacc[tc] = __builtin_amdgcn_wmma_f32_16x16x32_bf16(false, af, false, bfr,
                                                               (short)0, hacc[tc], false, false);
        }
    }
#pragma unroll
    for (int tc = 0; tc < 8; ++tc) {
#pragma unroll
        for (int j = 0; j < 8; ++j) {
            int t = wrow + (hf << 3) + j;
            int d = tc * 16 + r;
            hb[(size_t)(rowbase + t) * DIN + (h << 7) + d] = hacc[tc][j] * rden[t];
        }
    }
}

// ---------------------------------------------------------------------------
// Head-norm over DH + skip*xc + SiLU(z) gating. In-place on hb.
// ---------------------------------------------------------------------------
__global__ __launch_bounds__(128)
void headnorm_gate_kernel(float* __restrict__ hb, const float* __restrict__ xc,
                          const float* __restrict__ xz,
                          const float* __restrict__ hn_g,
                          const float* __restrict__ skip) {
    int rh = blockIdx.x;
    int row = rh >> 3, h = rh & 7;
    int tid = threadIdx.x;
    int c = (h << 7) + tid;
    float v = hb[(size_t)row * DIN + c];
    __shared__ float red[128];
    red[tid] = v; __syncthreads();
    for (int st = 64; st > 0; st >>= 1) { if (tid < st) red[tid] += red[tid + st]; __syncthreads(); }
    float mu = red[0] * (1.f / 128.f); __syncthreads();
    float d = v - mu;
    red[tid] = d * d; __syncthreads();
    for (int st = 64; st > 0; st >>= 1) { if (tid < st) red[tid] += red[tid + st]; __syncthreads(); }
    float var = red[0] * (1.f / 128.f); __syncthreads();
    float hn = d * rsqrtf(var + 1e-5f) * hn_g[c];
    hn += skip[c] * xc[(size_t)row * DIN + c];
    float z = xz[(size_t)row * 2048 + DIN + c];
    hb[(size_t)row * DIN + c] = hn * fast_silu(z);
}

// ---------------------------------------------------------------------------
// Host orchestration
// ---------------------------------------------------------------------------
static void gemm(hipStream_t st, const float* A, int lda, const float* B, int ldb,
                 const float* bias, const float* res, int ldr,
                 float* C, int ldc, int M, int N, int K, float alpha, int act) {
    dim3 g((N + 63) / 64, (M + 63) / 64);
    wmma_gemm_kernel<<<g, 256, 0, st>>>(A, lda, B, ldb, bias, res, ldr, C, ldc,
                                        M, N, K, alpha, act);
}

extern "C" void kernel_launch(void* const* d_in, const int* in_sizes, int n_in,
                              void* d_out, int out_size, void* d_ws, size_t ws_size,
                              hipStream_t stream) {
    (void)in_sizes; (void)n_in; (void)out_size; (void)ws_size;
#define IN(i) ((const float*)d_in[(i)])
    // JAX pytree (sorted keys): blocks[0] 0..16, blocks[1] 17..33, lv 34..39,
    // mu 40..45, out_b 46, out_g 47, x 48.
    float* ws   = (float*)d_ws;
    float* x_cur = ws;
    float* x_nxt = x_cur + 524288;
    float* lnb   = x_nxt + 524288;
    float* xz    = lnb   + 524288;
    float* xc    = xz    + 2097152;
    float* qb    = xc    + 1048576;
    float* kb    = qb    + 1048576;
    float* vb    = kb    + 1048576;
    float* ipb   = vb    + 1048576;
    float* fpb   = ipb   + 8192;
    float* hb    = fpb   + 8192;
    float* t1    = hb    + 1048576;
    float* t2    = t1    + 8192;

    hipMemcpyAsync(x_cur, IN(48), 524288 * sizeof(float),
                   hipMemcpyDeviceToDevice, stream);

    for (int p = 0; p < 2; ++p) {
        int bo = p * 17;
        const float* W_down = IN(bo + 0);
        const float* W_f    = IN(bo + 1);
        const float* W_i    = IN(bo + 2);
        const float* W_up   = IN(bo + 3);
        const float* Wk     = IN(bo + 4);
        const float* Wq     = IN(bo + 5);
        const float* Wv     = IN(bo + 6);
        const float* b_down = IN(bo + 7);
        const float* b_f    = IN(bo + 8);
        const float* b_i    = IN(bo + 9);
        const float* b_up   = IN(bo + 10);
        const float* conv_b = IN(bo + 11);
        const float* conv_w = IN(bo + 12);
        const float* hn_g   = IN(bo + 13);
        const float* ln_b   = IN(bo + 14);
        const float* ln_g   = IN(bo + 15);
        const float* skip   = IN(bo + 16);

        ln_kernel<<<NROWS, 256, 0, stream>>>(x_cur, ln_g, ln_b, lnb, DMODEL);
        gemm(stream, lnb, DMODEL, W_up, 2048, b_up, nullptr, 0,
             xz, 2048, NROWS, 2048, DMODEL, 1.f, 0);
        conv_silu_kernel<<<(NROWS * DIN) / 256, 256, 0, stream>>>(xz, conv_w, conv_b, xc);
        gemm(stream, xc, DIN, Wq, DIN, nullptr, nullptr, 0,
             qb, DIN, NROWS, DIN, DIN, 1.f, 0);
        gemm(stream, xc, DIN, Wk, DIN, nullptr, nullptr, 0,
             kb, DIN, NROWS, DIN, DIN, 0.08838834764831845f /* 1/sqrt(DH) */, 0);
        gemm(stream, xz, 2048, Wv, DIN, nullptr, nullptr, 0,
             vb, DIN, NROWS, DIN, DIN, 1.f, 0);
        gates_kernel<<<(NROWS * NHEAD * 32) / 256, 256, 0, stream>>>(
            xc, W_i, b_i, W_f, b_f, ipb, fpb);
        mlstm_scan_kernel<<<NBATCH * NHEAD, 256, 0, stream>>>(qb, kb, vb, ipb, fpb, hb);
        headnorm_gate_kernel<<<NROWS * NHEAD, 128, 0, stream>>>(hb, xc, xz, hn_g, skip);
        gemm(stream, hb, DIN, W_down, DMODEL, b_down, x_cur, DMODEL,
             x_nxt, DMODEL, NROWS, DMODEL, DIN, 1.f, 0);
        float* tmp = x_cur; x_cur = x_nxt; x_nxt = tmp;
    }

    ln_kernel<<<NROWS, 256, 0, stream>>>(x_cur, IN(47), IN(46), lnb, DMODEL);
    float* out = (float*)d_out;
    const int head_base[2] = {40 /* mu */, 34 /* lv */};
    const float* A0 = lnb + (SEQ - 1) * DMODEL;   // batch rows at stride SEQ*DMODEL
    for (int hh = 0; hh < 2; ++hh) {
        int hbn = head_base[hh];
        float* outp = out + hh * (NBATCH * 128);
        gemm(stream, A0, SEQ * DMODEL, IN(hbn + 0), 1024, IN(hbn + 3), nullptr, 0,
             t1, 1024, NBATCH, 1024, DMODEL, 1.f, 1);
        gemm(stream, t1, 1024, IN(hbn + 1), 512, IN(hbn + 4), nullptr, 0,
             t2, 512, NBATCH, 512, 1024, 1.f, 1);
        gemm(stream, t2, 512, IN(hbn + 2), 128, IN(hbn + 5), nullptr, 0,
             outp, 128, NBATCH, 128, 512, 1.f, 0);
    }
#undef IN
}
